// Decoder_84602265797072
// MI455X (gfx1250) — matrix operand order repaired and checked by reading, not compiled, not run on hardware
//
#include <hip/hip_runtime.h>
#include <hip/hip_bf16.h>

// ---------------- problem constants (from the reference) ----------------
#define PP      256
#define NATOMS  16
#define NN      4096      // P * NUM_ATOMS
#define HH      64
#define LL      16
#define LATENT  128
#define EE      65536
#define BB      2

typedef __attribute__((ext_vector_type(16))) _Float16 v16h;
typedef __attribute__((ext_vector_type(8)))  float    v8f;

// --------------------------- helpers ---------------------------

__device__ __forceinline__ float silu_f(float x) {
    return x / (1.0f + __expf(-x));
}

// Wave-private LDS producer->consumer sync (cross-lane, same wave).
// DS returns are in-order; s_wait_dscnt 0 guarantees the writes landed.
__device__ __forceinline__ void wave_lds_sync() {
    __builtin_amdgcn_wave_barrier();
    asm volatile("s_wait_dscnt 0" ::: "memory");
    __builtin_amdgcn_wave_barrier();
}

__device__ __forceinline__ v8f wmma16(v16h a, v16h b, v8f c) {
    // D = A(16x32 f16) * B(32x16 f16) + C(16x16 f32)
    return __builtin_amdgcn_wmma_f32_16x16x32_f16(
        false, a, false, b, (short)0, c, false, false);
}

// A fragment (16x32 f16) from an LDS/global row-major tile, row stride `stride`
// halves, K-chunk kc (covers K = kc*32 .. kc*32+31).
// Lane l: row M = l&15; lanes 0-15 hold K{0..7,16..23}, lanes 16-31 K{8..15,24..31}.
__device__ __forceinline__ v16h load_A_rows(const _Float16* t, int stride,
                                            int lane, int kc) {
    const int row = lane & 15;
    const int kb  = kc * 32 + ((lane >> 4) << 3);
    const _Float16* p = t + row * stride + kb;
    v16h a;
#pragma unroll
    for (int i = 0; i < 8; ++i) { a[i] = p[i]; a[i + 8] = p[16 + i]; }
    return a;
}

// Same A layout but the per-row base pointers differ (gathered rows).
__device__ __forceinline__ v16h load_A_ptr(const _Float16* rowp, int lane, int kc) {
    const int kb = (kc & 1) * 32 + ((lane >> 4) << 3);
    const _Float16* p = rowp + kb;
    v16h a;
#pragma unroll
    for (int i = 0; i < 8; ++i) { a[i] = p[i]; a[i + 8] = p[16 + i]; }
    return a;
}

// B fragment (32x16 f16) from an N-major f16 weight matrix Wt[n][k] (stride K).
// Lane l: col N = (l&15)+nt*16; lanes 0-15 hold K{0..7,16..23}, lanes 16-31 K{8..15,24..31}.
__device__ __forceinline__ v16h load_B_nmajor(const _Float16* Wt, int K,
                                              int lane, int kc, int nt) {
    const int n  = nt * 16 + (lane & 15);
    const int k0 = kc * 32 + ((lane >> 4) << 3);
    const _Float16* p = Wt + (size_t)n * K + k0;
    v16h b;
#pragma unroll
    for (int i = 0; i < 8; ++i) { b[i] = p[i]; b[i + 8] = p[16 + i]; }
    return b;
}

// --------------------------- prep kernels ---------------------------

// dst[l][n][k] = (f16) src[l][k][n]   (transpose+convert weights, N-major)
__global__ void k_convert_wt(const float* __restrict__ src,
                             _Float16* __restrict__ dst,
                             int K, int Ncols, int layers) {
    int idx = blockIdx.x * blockDim.x + threadIdx.x;
    int total = layers * K * Ncols;
    if (idx >= total) return;
    int l   = idx / (K * Ncols);
    int rem = idx % (K * Ncols);
    int k   = rem / Ncols;
    int n   = rem % Ncols;
    dst[(size_t)l * Ncols * K + (size_t)n * K + k] = (_Float16)src[idx];
}

// z_cond[b][h] = z[b] @ deconv_w[:,h] + deconv_b[h]
__global__ void k_zcond(const float* __restrict__ z,
                        const float* __restrict__ dw,
                        const float* __restrict__ db,
                        float* __restrict__ zc) {
    int idx = blockIdx.x * blockDim.x + threadIdx.x;   // BB*HH threads
    if (idx >= BB * HH) return;
    int b = idx / HH, h = idx % HH;
    float s = db[h];
#pragma unroll 4
    for (int k = 0; k < LATENT; ++k) s += z[b * LATENT + k] * dw[k * HH + h];
    zc[idx] = s;
}

// h0 = feat @ emb_w + emb_b + z_cond ; also zero agg, mirror to f16
__global__ void k_init_h(const float* __restrict__ emb_w,
                         const float* __restrict__ emb_b,
                         const float* __restrict__ zc,
                         float* __restrict__ h,
                         _Float16* __restrict__ hf,
                         float* __restrict__ agg) {
    int idx = blockIdx.x * blockDim.x + threadIdx.x;   // BB*NN*HH, exact
    int b    = idx / (NN * HH);
    int rem  = idx % (NN * HH);
    int node = rem / HH;
    int hh   = rem % HH;
    float p = (float)(node / NATOMS) * (1.0f / (PP - 1));
    float a = (float)(node % NATOMS) * (1.0f / (NATOMS - 1));
    float v = p * emb_w[hh] + a * emb_w[HH + hh] + emb_w[2 * HH + hh]
            + emb_b[hh] + zc[b * HH + hh];
    h[idx]   = v;
    hf[idx]  = (_Float16)v;
    agg[idx] = 0.0f;
}

// pos = broadcast(initial_pos) ; zero pos_delta. pos lives in d_out.
__global__ void k_init_pos(const float* __restrict__ ipos,
                           float* __restrict__ pos,
                           float* __restrict__ pdelta) {
    int idx = blockIdx.x * blockDim.x + threadIdx.x;   // BB*NN*3, exact
    int r = idx % (NN * 3);
    pos[idx]    = ipos[r];
    pdelta[idx] = 0.0f;
}

// --------------------------- edge kernel ---------------------------
// One wave = 16 edges. 3 WMMA matmuls (K=128(+1), 64, 64) + w-dot + scatters.
__global__ __launch_bounds__(256) void k_edge(
    const int* __restrict__ ei,          // [2*E]
    const float* __restrict__ pos,       // [B][N][3]  (== d_out)
    const _Float16* __restrict__ hf,     // [B][N][H]
    const _Float16* __restrict__ ew1t,   // [L][64][129]
    const float* __restrict__ eb1,       // [L][64]
    const _Float16* __restrict__ ew2t,   // [L][64][64]
    const float* __restrict__ eb2,       // [L][64]
    const _Float16* __restrict__ cw1t,   // [L][64][64]
    const float* __restrict__ cb1,       // [L][64]
    const _Float16* __restrict__ cw2h,   // [L][64]
    const float* __restrict__ cb2,       // [L][1]
    float* __restrict__ agg,             // [B][N][H]
    float* __restrict__ pdelta,          // [B][N][3]
    int layer)
{
    __shared__ _Float16 tA[8][16 * 64];
    __shared__ _Float16 tB[8][16 * 64];
    __shared__ float    sdif[8][16][3];
    __shared__ float    sd2[8][16];
    __shared__ int      srow[8][16];

    const int wid  = threadIdx.x >> 5;
    const int lane = threadIdx.x & 31;
    const int tile = blockIdx.x * 8 + wid;           // 8192 tiles, exact grid
    const int tiles_per_b = EE / 16;
    const int b     = tile / tiles_per_b;
    const int ebase = (tile % tiles_per_b) * 16;

    const _Float16* W1 = ew1t + (size_t)layer * 64 * 129;
    const _Float16* W2 = ew2t + (size_t)layer * 64 * 64;
    const _Float16* C1 = cw1t + (size_t)layer * 64 * 64;
    const _Float16* C2 = cw2h + (size_t)layer * 64;
    const float* B1 = eb1 + layer * 64;
    const float* B2 = eb2 + layer * 64;
    const float* BC1 = cb1 + layer * 64;
    const float  bc2 = cb2[layer];

    // Per-lane edge row (row M = lane&15 of the tile)
    const int er = lane & 15;
    const int r  = ei[ebase + er];
    const int c  = ei[EE + ebase + er];

    if (lane < 16) {
        const float* pr = pos + ((size_t)b * NN + r) * 3;
        const float* pc = pos + ((size_t)b * NN + c) * 3;
        float dx = pr[0] - pc[0], dy = pr[1] - pc[1], dz = pr[2] - pc[2];
        sdif[wid][er][0] = dx; sdif[wid][er][1] = dy; sdif[wid][er][2] = dz;
        sd2[wid][er] = dx * dx + dy * dy + dz * dz;
        srow[wid][er] = r;
    }
    wave_lds_sync();

    const v8f vz = {0.f, 0.f, 0.f, 0.f, 0.f, 0.f, 0.f, 0.f};
    v8f acc[4];

    // ---- matmul 1: [hi | hj](16x128) @ ew1[0:128]  (+ d2 * ew1[128] + b1)
#pragma unroll
    for (int nt = 0; nt < 4; ++nt) acc[nt] = vz;
#pragma unroll
    for (int kc = 0; kc < 4; ++kc) {
        const int node = (kc < 2) ? r : c;
        const _Float16* rowp = hf + ((size_t)b * NN + node) * HH;
        v16h afr = load_A_ptr(rowp, lane, kc);
#pragma unroll
        for (int nt = 0; nt < 4; ++nt) {
            v16h bfr = load_B_nmajor(W1, 129, lane, kc, nt);
            acc[nt] = wmma16(afr, bfr, acc[nt]);
        }
    }
#pragma unroll
    for (int nt = 0; nt < 4; ++nt) {
        const int n = nt * 16 + (lane & 15);
        const float w129 = (float)W1[(size_t)n * 129 + 128];
        const float bias = B1[n];
#pragma unroll
        for (int i = 0; i < 8; ++i) {
            const int m = ((lane >> 4) << 3) + i;
            float x = acc[nt][i] + bias + sd2[wid][m] * w129;
            tA[wid][m * 64 + n] = (_Float16)silu_f(x);
        }
    }
    wave_lds_sync();

    // ---- matmul 2: m2 = silu(m1 @ ew2 + b2); scatter m2 into agg
#pragma unroll
    for (int nt = 0; nt < 4; ++nt) acc[nt] = vz;
#pragma unroll
    for (int kc = 0; kc < 2; ++kc) {
        v16h afr = load_A_rows(tA[wid], 64, lane, kc);
#pragma unroll
        for (int nt = 0; nt < 4; ++nt) {
            v16h bfr = load_B_nmajor(W2, 64, lane, kc, nt);
            acc[nt] = wmma16(afr, bfr, acc[nt]);
        }
    }
#pragma unroll
    for (int nt = 0; nt < 4; ++nt) {
        const int n = nt * 16 + (lane & 15);
        const float bias = B2[n];
#pragma unroll
        for (int i = 0; i < 8; ++i) {
            const int m = ((lane >> 4) << 3) + i;
            float x = silu_f(acc[nt][i] + bias);
            tB[wid][m * 64 + n] = (_Float16)x;
            atomicAdd(&agg[((size_t)b * NN + srow[wid][m]) * HH + n], x);
        }
    }
    wave_lds_sync();

    // ---- matmul 3: t = silu(m2 @ cw1 + cb1)
#pragma unroll
    for (int nt = 0; nt < 4; ++nt) acc[nt] = vz;
#pragma unroll
    for (int kc = 0; kc < 2; ++kc) {
        v16h afr = load_A_rows(tB[wid], 64, lane, kc);
#pragma unroll
        for (int nt = 0; nt < 4; ++nt) {
            v16h bfr = load_B_nmajor(C1, 64, lane, kc, nt);
            acc[nt] = wmma16(afr, bfr, acc[nt]);
        }
    }
#pragma unroll
    for (int nt = 0; nt < 4; ++nt) {
        const int n = nt * 16 + (lane & 15);
        const float bias = BC1[n];
#pragma unroll
        for (int i = 0; i < 8; ++i) {
            const int m = ((lane >> 4) << 3) + i;
            tA[wid][m * 64 + n] = (_Float16)silu_f(acc[nt][i] + bias);
        }
    }
    wave_lds_sync();

    // ---- w = t @ cw2 + cb2 ; pos_delta += diff * w
    if (lane < 16) {
        float ws = bc2;
#pragma unroll 8
        for (int k = 0; k < 64; ++k)
            ws += (float)tA[wid][er * 64 + k] * (float)C2[k];
        float* pd = pdelta + ((size_t)b * NN + srow[wid][er]) * 3;
        atomicAdd(&pd[0], sdif[wid][er][0] * ws);
        atomicAdd(&pd[1], sdif[wid][er][1] * ws);
        atomicAdd(&pd[2], sdif[wid][er][2] * ws);
    }
}

// --------------------------- node kernel ---------------------------
// One wave = 16 nodes: upd = silu([h|agg] @ nw1 + nb1) @ nw2 + nb2 ; h += upd
// Also applies pos += pos_delta and zeroes agg/pos_delta for the next layer.
__global__ __launch_bounds__(256) void k_node(
    float* __restrict__ h,               // [B][N][H]
    _Float16* __restrict__ hf,           // [B][N][H]
    float* __restrict__ agg,             // [B][N][H]
    float* __restrict__ pos,             // [B][N][3]
    float* __restrict__ pdelta,          // [B][N][3]
    const _Float16* __restrict__ nw1t,   // [L][64][128]
    const float* __restrict__ nb1,       // [L][64]
    const _Float16* __restrict__ nw2t,   // [L][64][64]
    const float* __restrict__ nb2,       // [L][64]
    int layer)
{
    __shared__ _Float16 tIn[8][16 * 128];
    __shared__ _Float16 tMid[8][16 * 64];

    const int wid  = threadIdx.x >> 5;
    const int lane = threadIdx.x & 31;
    const int tile = blockIdx.x * 8 + wid;          // 512 tiles, exact grid
    const int tiles_per_b = NN / 16;
    const int b     = tile / tiles_per_b;
    const int nbase = (tile % tiles_per_b) * 16;

    const _Float16* W1 = nw1t + (size_t)layer * 64 * 128;
    const _Float16* W2 = nw2t + (size_t)layer * 64 * 64;
    const float* B1 = nb1 + layer * 64;
    const float* B2 = nb2 + layer * 64;

    // stage [h | agg] as f16, zero agg behind us
    for (int idx = lane; idx < 16 * 128; idx += 32) {
        const int m = idx >> 7;
        const int k = idx & 127;
        const size_t base = ((size_t)b * NN + nbase + m) * HH;
        float v;
        if (k < 64) v = h[base + k];
        else { v = agg[base + k - 64]; agg[base + k - 64] = 0.0f; }
        tIn[wid][idx] = (_Float16)v;
    }
    wave_lds_sync();

    const v8f vz = {0.f, 0.f, 0.f, 0.f, 0.f, 0.f, 0.f, 0.f};
    v8f acc[4];

    // ---- matmul 1 (K = 128)
#pragma unroll
    for (int nt = 0; nt < 4; ++nt) acc[nt] = vz;
#pragma unroll
    for (int kc = 0; kc < 4; ++kc) {
        v16h afr = load_A_rows(tIn[wid], 128, lane, kc);
#pragma unroll
        for (int nt = 0; nt < 4; ++nt) {
            v16h bfr = load_B_nmajor(W1, 128, lane, kc, nt);
            acc[nt] = wmma16(afr, bfr, acc[nt]);
        }
    }
#pragma unroll
    for (int nt = 0; nt < 4; ++nt) {
        const int n = nt * 16 + (lane & 15);
        const float bias = B1[n];
#pragma unroll
        for (int i = 0; i < 8; ++i) {
            const int m = ((lane >> 4) << 3) + i;
            tMid[wid][m * 64 + n] = (_Float16)silu_f(acc[nt][i] + bias);
        }
    }
    wave_lds_sync();

    // ---- matmul 2 (K = 64); residual add into h
#pragma unroll
    for (int nt = 0; nt < 4; ++nt) acc[nt] = vz;
#pragma unroll
    for (int kc = 0; kc < 2; ++kc) {
        v16h afr = load_A_rows(tMid[wid], 64, lane, kc);
#pragma unroll
        for (int nt = 0; nt < 4; ++nt) {
            v16h bfr = load_B_nmajor(W2, 64, lane, kc, nt);
            acc[nt] = wmma16(afr, bfr, acc[nt]);
        }
    }
#pragma unroll
    for (int nt = 0; nt < 4; ++nt) {
        const int n = nt * 16 + (lane & 15);
        const float bias = B2[n];
#pragma unroll
        for (int i = 0; i < 8; ++i) {
            const int m = ((lane >> 4) << 3) + i;
            const size_t hidx = ((size_t)b * NN + nbase + m) * HH + n;
            float hv = h[hidx] + acc[nt][i] + bias;
            h[hidx]  = hv;
            hf[hidx] = (_Float16)hv;
        }
    }

    // pos += pos_delta ; pos_delta = 0
    if (lane < 16) {
        const size_t pbase = ((size_t)b * NN + nbase + lane) * 3;
#pragma unroll
        for (int cc = 0; cc < 3; ++cc) {
            pos[pbase + cc] += pdelta[pbase + cc];
            pdelta[pbase + cc] = 0.0f;
        }
    }
}

// --------------------------- launcher ---------------------------

extern "C" void kernel_launch(void* const* d_in, const int* in_sizes, int n_in,
                              void* d_out, int out_size, void* d_ws, size_t ws_size,
                              hipStream_t stream) {
    const float* z        = (const float*)d_in[0];
    const int*   ei       = (const int*)d_in[1];     // [2][E]
    const float* ipos     = (const float*)d_in[2];   // [1][N][3]
    const float* deconv_w = (const float*)d_in[3];
    const float* deconv_b = (const float*)d_in[4];
    const float* emb_w    = (const float*)d_in[5];
    const float* emb_b    = (const float*)d_in[6];
    const float* edge_w1  = (const float*)d_in[7];
    const float* edge_b1  = (const float*)d_in[8];
    const float* edge_w2  = (const float*)d_in[9];
    const float* edge_b2  = (const float*)d_in[10];
    const float* coord_w1 = (const float*)d_in[11];
    const float* coord_b1 = (const float*)d_in[12];
    const float* coord_w2 = (const float*)d_in[13];
    const float* coord_b2 = (const float*)d_in[14];
    const float* node_w1  = (const float*)d_in[15];
    const float* node_b1  = (const float*)d_in[16];
    const float* node_w2  = (const float*)d_in[17];
    const float* node_b2  = (const float*)d_in[18];

    float* pos = (float*)d_out;                      // [B][N][3] == output

    // workspace carve-up
    char* ws = (char*)d_ws;
    size_t off = 0;
    auto carve = [&](size_t bytes) -> void* {
        void* p = ws + off;
        off = (off + bytes + 255) & ~(size_t)255;
        return p;
    };
    float*    h      = (float*)   carve((size_t)BB * NN * HH * 4);
    _Float16* hfp    = (_Float16*)carve((size_t)BB * NN * HH * 2);
    float*    agg    = (float*)   carve((size_t)BB * NN * HH * 4);
    float*    pdelta = (float*)   carve((size_t)BB * NN * 3 * 4);
    float*    zc     = (float*)   carve((size_t)BB * HH * 4);
    _Float16* ew1t   = (_Float16*)carve((size_t)LL * 64 * 129 * 2);
    _Float16* ew2t   = (_Float16*)carve((size_t)LL * 64 * 64 * 2);
    _Float16* cw1t   = (_Float16*)carve((size_t)LL * 64 * 64 * 2);
    _Float16* cw2h   = (_Float16*)carve((size_t)LL * 64 * 2);
    _Float16* nw1t   = (_Float16*)carve((size_t)LL * 64 * 128 * 2);
    _Float16* nw2t   = (_Float16*)carve((size_t)LL * 64 * 64 * 2);

    // ---- weight conversion (transpose to N-major f16)
    auto cvt = [&](const float* src, _Float16* dst, int K, int Ncols) {
        int total = LL * K * Ncols;
        k_convert_wt<<<(total + 255) / 256, 256, 0, stream>>>(src, dst, K, Ncols, LL);
    };
    cvt(edge_w1,  ew1t, 129, 64);
    cvt(edge_w2,  ew2t,  64, 64);
    cvt(coord_w1, cw1t,  64, 64);
    cvt(coord_w2, cw2h,  64,  1);
    cvt(node_w1,  nw1t, 128, 64);
    cvt(node_w2,  nw2t,  64, 64);

    // ---- init
    k_zcond<<<1, 128, 0, stream>>>(z, deconv_w, deconv_b, zc);
    k_init_h<<<(BB * NN * HH) / 256, 256, 0, stream>>>(emb_w, emb_b, zc, h, hfp, agg);
    k_init_pos<<<(BB * NN * 3) / 256, 256, 0, stream>>>(ipos, pos, pdelta);

    // ---- L message-passing layers
    const int edge_blocks = (EE / 16) * BB / 8;   // 1024, exact (full waves)
    const int node_blocks = (NN / 16) * BB / 8;   // 64, exact (full waves)
    for (int l = 0; l < LL; ++l) {
        k_edge<<<edge_blocks, 256, 0, stream>>>(
            ei, pos, hfp,
            ew1t, edge_b1, ew2t, edge_b2,
            cw1t, coord_b1, cw2h, coord_b2,
            agg, pdelta, l);
        k_node<<<node_blocks, 256, 0, stream>>>(
            h, hfp, agg, pos, pdelta,
            nw1t, node_b1, nw2t, node_b2, l);
    }
}
